// EncoderDecoderRecurrentCell_78993038508598
// MI455X (gfx1250) — matrix operand order repaired and checked by reading, not compiled, not run on hardware
//
#include <hip/hip_runtime.h>
#include <hip/hip_bf16.h>
#include <math.h>

typedef __bf16 bf16_t;
typedef __attribute__((ext_vector_type(16))) __bf16 v16bf;
typedef __attribute__((ext_vector_type(8)))  float   v8f;

// Problem dimensions (compile-time, from reference)
#define cB    64      // batch
#define cT    1024    // past timesteps
#define cFIN  64      // input features
#define cRI   256     // recurrent input size
#define cH    512     // hidden
#define cFOUT 64      // output features
#define cTT   1280    // total steps (past + future)
#define cG    2048    // 4*H gate columns
#define NWG   16      // persistent workgroups in the recurrent kernel

// LDS layout of the recurrent kernel (dynamic shared, base offset 0)
#define LDS_WIH_OFF 0                          // 128 rows x 256 bf16 = 64 KB
#define LDS_WHH_OFF (128 * 256 * 2)            // 128 rows x 512 bf16 = 128 KB
#define LDS_X_OFF   (LDS_WHH_OFF + 128 * 512 * 2)
#define LDS_XBUF    (cB * cRI * 2)             // 32 KB per x buffer
#define LDS_TOTAL   (LDS_X_OFF + 2 * LDS_XBUF) // 256 KB total

// ---------------------------------------------------------------- helpers ---

static __device__ __forceinline__ v8f v8f_zero() {
  v8f v;
#pragma unroll
  for (int i = 0; i < 8; ++i) v[i] = 0.0f;
  return v;
}

static __device__ __forceinline__ v8f v8f_splat(float x) {
  v8f v;
#pragma unroll
  for (int i = 0; i < 8; ++i) v[i] = x;
  return v;
}

// A fragment: 16x32 bf16 tile, row-major source with leading dim ld.
// lane holds row (lane&15); K chunks {kb..kb+7, kb+16..kb+23}, kb = lane<16?0:8
static __device__ __forceinline__ v16bf load_fragA(const bf16_t* p0, int ld) {
  const int lane = threadIdx.x & 31;
  const bf16_t* p = p0 + (size_t)(lane & 15) * ld + ((lane < 16) ? 0 : 8);
  union { v16bf v; uint4 q[2]; } u;
  u.q[0] = *(const uint4*)(p);
  u.q[1] = *(const uint4*)(p + 16);
  return u.v;
}

// B fragment: 32x16 bf16 tile from N-major weights (W[n][k], leading dim ld).
// lane holds column n=(lane&15); K chunks {kb..kb+7, kb+16..kb+23}, kb = lane<16?0:8
static __device__ __forceinline__ v16bf load_fragB(const bf16_t* p0, int ld) {
  const int lane = threadIdx.x & 31;
  const bf16_t* p = p0 + (size_t)(lane & 15) * ld + ((lane < 16) ? 0 : 8);
  union { v16bf v; uint4 q[2]; } u;
  u.q[0] = *(const uint4*)(p);
  u.q[1] = *(const uint4*)(p + 16);
  return u.v;
}

static __device__ __forceinline__ v8f wmma_bf16(v16bf a, v16bf b, v8f c) {
  return __builtin_amdgcn_wmma_f32_16x16x32_bf16(false, a, false, b, (short)0, c,
                                                 false, false);
}

static __device__ __forceinline__ float sigm(float x) {
  return 1.0f / (1.0f + __expf(-x));
}

// CDNA5 async global->LDS copy of 16 bytes (ASYNCcnt tracked).
static __device__ __forceinline__ void async_copy16(unsigned lds_byte_off,
                                                    const void* gp) {
  unsigned long long ga = (unsigned long long)gp;
  asm volatile("global_load_async_to_lds_b128 %0, %1, off"
               :: "v"(lds_byte_off), "v"(ga)
               : "memory");
}

static __device__ __forceinline__ void wait_async0() {
  asm volatile("s_wait_asynccnt 0x0" ::: "memory");
}

// Grid-wide sense barrier for the persistent recurrent kernel.
static __device__ __forceinline__ void grid_barrier(int* cnt, int* gen) {
  __threadfence();          // flush this thread's h stores to device scope
  __syncthreads();
  if (threadIdx.x == 0) {
    volatile int* vgen = (volatile int*)gen;
    int g = *vgen;
    if (atomicAdd(cnt, 1) == NWG - 1) {
      *(volatile int*)cnt = 0;
      __threadfence();
      atomicAdd(gen, 1);
    } else {
      while (*vgen == g) { __builtin_amdgcn_s_sleep(1); }
    }
  }
  __syncthreads();
  __threadfence();          // acquire: see other WGPs' h stores
}

// ------------------------------------------------------------ prep kernels ---

__global__ void k_zero_i32(int* p, int n) {
  int i = blockIdx.x * blockDim.x + threadIdx.x;
  if (i < n) p[i] = 0;
}

__global__ void k_cast_bf16(const float* __restrict__ s, bf16_t* __restrict__ d, int n) {
  int i = blockIdx.x * blockDim.x + threadIdx.x;
  if (i < n) d[i] = (bf16_t)s[i];
}

// src[R][C] f32 -> dst[C][R] bf16 (make weights N-major for B fragments)
__global__ void k_tr_cast(const float* __restrict__ s, bf16_t* __restrict__ d,
                          int R, int C) {
  int i = blockIdx.x * blockDim.x + threadIdx.x;
  if (i < R * C) {
    int r = i / C, c = i - r * C;
    d[(size_t)c * R + r] = (bf16_t)s[i];
  }
}

// input_seq [b][t][k] f32 -> in_bf [t][b][k] bf16 (batch-major rows per step)
__global__ void k_in_tr(const float* __restrict__ s, bf16_t* __restrict__ d) {
  int i = blockIdx.x * blockDim.x + threadIdx.x;
  if (i < cB * cT * cFIN) {
    int b = i / (cT * cFIN);
    int rem = i - b * (cT * cFIN);
    int t = rem / cFIN;
    int k = rem - t * cFIN;
    d[(size_t)t * (cB * cFIN) + b * cFIN + k] = (bf16_t)s[i];
  }
}

__global__ void k_fill_bf16(bf16_t* d, int n, float v) {
  int i = blockIdx.x * blockDim.x + threadIdx.x;
  if (i < n) d[i] = (bf16_t)v;
}

// bias_p[n] = b_ih + b_hh ; bias_f[n] = bias_p[n] + W_ih[n,:]·fut_b
__global__ void k_bias(const float* __restrict__ b_ih, const float* __restrict__ b_hh,
                       const float* __restrict__ W_ih, const float* __restrict__ fut_b,
                       float* __restrict__ bp, float* __restrict__ bfu) {
  int n = blockIdx.x * blockDim.x + threadIdx.x;
  if (n < cG) {
    float p = b_ih[n] + b_hh[n];
    float s = 0.0f;
    for (int m = 0; m < cRI; ++m) s += W_ih[(size_t)n * cRI + m] * fut_b[m];
    bp[n] = p;
    bfu[n] = p + s;
  }
}

// W_comb[n][k] = W_hh[n][k] + sum_m W_ih[n][m] * fut_W[k][m]   (2048x512 bf16)
__global__ void k_comb(const float* __restrict__ W_hh, const float* __restrict__ W_ih,
                       const float* __restrict__ fut_W, bf16_t* __restrict__ Wc) {
  int i = blockIdx.x * blockDim.x + threadIdx.x;
  if (i < cG * cH) {
    int n = i >> 9;        // / 512
    int k = i & 511;
    float s = W_hh[(size_t)n * cH + k];
    for (int m = 0; m < cRI; ++m)
      s += W_ih[(size_t)n * cRI + m] * fut_W[(size_t)k * cRI + m];
    Wc[i] = (bf16_t)s;
  }
}

// ------------------------------------------------------ encoder GEMM (WMMA) ---
// per block: one timestep t.  M=64(batch) x N=256 x K=64, bf16 WMMA.
__global__ void __launch_bounds__(256) k_encoder(const bf16_t* __restrict__ in_bf,
                                                 const bf16_t* __restrict__ encWT,
                                                 const float* __restrict__ enc_b,
                                                 bf16_t* __restrict__ enc_bf) {
  const int t = blockIdx.x;
  const int wave = threadIdx.x >> 5, lane = threadIdx.x & 31;
  const int n = lane & 15, mb = (lane < 16) ? 0 : 8;
  const bf16_t* A0 = in_bf + (size_t)t * (cB * cFIN);
  bf16_t* O = enc_bf + (size_t)t * (cB * cRI);
  for (int tid = wave; tid < 64; tid += 8) {       // 4 mtiles x 16 ntiles
    int m0 = (tid & 3) * 16, n0 = (tid >> 2) * 16;
    v8f acc = v8f_zero();
    for (int k = 0; k < cFIN; k += 32) {
      v16bf a = load_fragA(A0 + (size_t)m0 * cFIN + k, cFIN);
      v16bf b = load_fragB(encWT + (size_t)n0 * cFIN + k, cFIN);
      acc = wmma_bf16(a, b, acc);
    }
    float bias = enc_b[n0 + n];
#pragma unroll
    for (int r = 0; r < 8; ++r) {
      int m = m0 + mb + r;
      O[(size_t)m * cRI + n0 + n] = (bf16_t)(acc[r] + bias);
    }
  }
}

// ----------------------------------------------- persistent recurrent (WMMA) ---
// 16 WGs x 8 waves, one WG per WGP (256 KB LDS each). WG owns 32 hidden units
// (all 4 gates). Wave = (mtile, ntile): 4 batch 16-rows x 2 hidden 16-cols.
// Weights staged in LDS once; x tiles double-buffered via async global->LDS;
// c kept in registers for all 1280 steps.
__global__ void __launch_bounds__(256) k_recurrent(
    const bf16_t* __restrict__ enc_bf, const bf16_t* __restrict__ h0_bf,
    const bf16_t* __restrict__ Wih, const bf16_t* __restrict__ Whh,
    const bf16_t* __restrict__ Wcmb,
    const float* __restrict__ bias_p, const float* __restrict__ bias_f,
    bf16_t* __restrict__ h_hist, int* bar_cnt, int* bar_gen) {
  extern __shared__ char smem[];
  bf16_t* lWih = (bf16_t*)(smem + LDS_WIH_OFF);   // [128][256]
  bf16_t* lWhh = (bf16_t*)(smem + LDS_WHH_OFF);   // [128][512] (Wcmb in future)

  const int wg = blockIdx.x;
  const int wave = threadIdx.x >> 5, lane = threadIdx.x & 31;
  const int m0 = (wave & 3) * 16;                 // batch tile base
  const int nt = (wave >> 2);                     // 0..1 local hidden tile
  const int jb = wg * 32 + nt * 16;               // global hidden-column base
  const int n = lane & 15, mb = (lane < 16) ? 0 : 8;

  // ---- one-time stage: W_ih slice (128 rows x 256) and W_hh slice (128 x 512)
  for (int idx = threadIdx.x; idx < 128 * 32; idx += 256) {
    int r = idx >> 5, ch = idx & 31;              // row, 16B-chunk
    int grow = (r >> 5) * cH + wg * 32 + (r & 31);
    *(uint4*)(lWih + r * cRI + ch * 8) =
        *(const uint4*)(Wih + (size_t)grow * cRI + ch * 8);
  }
  for (int idx = threadIdx.x; idx < 128 * 64; idx += 256) {
    int r = idx >> 6, ch = idx & 63;
    int grow = (r >> 5) * cH + wg * 32 + (r & 31);
    *(uint4*)(lWhh + r * cH + ch * 8) =
        *(const uint4*)(Whh + (size_t)grow * cH + ch * 8);
  }
  // ---- async prefetch x(0) into buffer 0 (8 x 16B per thread = 32 KB total)
  {
    const char* g = (const char*)enc_bf;
#pragma unroll
    for (int k2 = 0; k2 < 8; ++k2) {
      unsigned o = (unsigned)(threadIdx.x + k2 * 256) * 16u;
      async_copy16((unsigned)LDS_X_OFF + o, g + o);
    }
  }
  __syncthreads();

  // per-wave bias scalars, hoisted out of the time loop
  float bpv[4], bfv[4];
#pragma unroll
  for (int g = 0; g < 4; ++g) {
    bpv[g] = bias_p[g * cH + jb + n];
    bfv[g] = bias_f[g * cH + jb + n];
  }

  v8f c = v8f_splat(0.01f);                       // LSTM cell state, in registers
  const bf16_t* hprev = h0_bf;

  for (int t = 0; t < cTT; ++t) {
    v8f g0 = v8f_zero(), g1 = v8f_zero(), g2 = v8f_zero(), g3 = v8f_zero();
    const int rb = nt * 16;                       // local B row base
    if (t < cT) {
      // x(t) is in LDS buffer t&1 (async prefetch from step t-1); drain + sync.
      wait_async0();
      __syncthreads();
      // kick prefetch of x(t+1) into the other buffer, overlapping compute
      if (t + 1 < cT) {
        const char* g = (const char*)(enc_bf + (size_t)(t + 1) * (cB * cRI));
        unsigned lb = (unsigned)LDS_X_OFF + (unsigned)(((t + 1) & 1) * LDS_XBUF);
#pragma unroll
        for (int k2 = 0; k2 < 8; ++k2) {
          unsigned o = (unsigned)(threadIdx.x + k2 * 256) * 16u;
          async_copy16(lb + o, g + o);
        }
      }
      // gates = x@W_ih^T  (K=256; A from LDS, B from LDS)
      const bf16_t* A1 =
          (const bf16_t*)(smem + LDS_X_OFF + (size_t)(t & 1) * LDS_XBUF) +
          (size_t)m0 * cRI;
      for (int k = 0; k < cRI; k += 32) {
        v16bf a = load_fragA(A1 + k, cRI);
        g0 = wmma_bf16(a, load_fragB(lWih + (size_t)(0 * 32 + rb) * cRI + k, cRI), g0);
        g1 = wmma_bf16(a, load_fragB(lWih + (size_t)(1 * 32 + rb) * cRI + k, cRI), g1);
        g2 = wmma_bf16(a, load_fragB(lWih + (size_t)(2 * 32 + rb) * cRI + k, cRI), g2);
        g3 = wmma_bf16(a, load_fragB(lWih + (size_t)(3 * 32 + rb) * cRI + k, cRI), g3);
      }
      //        + h@W_hh^T  (K=512; A from global/L2, B from LDS)
      const bf16_t* A2 = hprev + (size_t)m0 * cH;
      for (int k = 0; k < cH; k += 32) {
        v16bf a = load_fragA(A2 + k, cH);
        g0 = wmma_bf16(a, load_fragB(lWhh + (size_t)(0 * 32 + rb) * cH + k, cH), g0);
        g1 = wmma_bf16(a, load_fragB(lWhh + (size_t)(1 * 32 + rb) * cH + k, cH), g1);
        g2 = wmma_bf16(a, load_fragB(lWhh + (size_t)(2 * 32 + rb) * cH + k, cH), g2);
        g3 = wmma_bf16(a, load_fragB(lWhh + (size_t)(3 * 32 + rb) * cH + k, cH), g3);
      }
    } else {
      if (t == cT) {
        // swap W_hh slice for fused future weights W_comb (one-time)
        __syncthreads();
        for (int idx = threadIdx.x; idx < 128 * 64; idx += 256) {
          int r = idx >> 6, ch = idx & 63;
          int grow = (r >> 5) * cH + wg * 32 + (r & 31);
          *(uint4*)(lWhh + r * cH + ch * 8) =
              *(const uint4*)(Wcmb + (size_t)grow * cH + ch * 8);
        }
        __syncthreads();
      }
      // future, fused: gates = h@(W_hh + W_ih@fut_W^T)^T  (K=512 only)
      const bf16_t* A2 = hprev + (size_t)m0 * cH;
      for (int k = 0; k < cH; k += 32) {
        v16bf a = load_fragA(A2 + k, cH);
        g0 = wmma_bf16(a, load_fragB(lWhh + (size_t)(0 * 32 + rb) * cH + k, cH), g0);
        g1 = wmma_bf16(a, load_fragB(lWhh + (size_t)(1 * 32 + rb) * cH + k, cH), g1);
        g2 = wmma_bf16(a, load_fragB(lWhh + (size_t)(2 * 32 + rb) * cH + k, cH), g2);
        g3 = wmma_bf16(a, load_fragB(lWhh + (size_t)(3 * 32 + rb) * cH + k, cH), g3);
      }
    }

    const bool past = (t < cT);
    const float bi = past ? bpv[0] : bfv[0];
    const float bf = past ? bpv[1] : bfv[1];
    const float bg = past ? bpv[2] : bfv[2];
    const float bo = past ? bpv[3] : bfv[3];
    bf16_t* hnext = h_hist + (size_t)t * (cB * cH);
#pragma unroll
    for (int r = 0; r < 8; ++r) {
      float iv = sigm(g0[r] + bi);
      float fv = sigm(g1[r] + bf);
      float gv = tanhf(g2[r] + bg);
      float ov = sigm(g3[r] + bo);
      float cn = fv * c[r] + iv * gv;
      c[r] = cn;
      float hv = ov * tanhf(cn);
      int m = m0 + mb + r;                        // batch row
      hnext[(size_t)m * cH + jb + n] = (bf16_t)hv;
    }

    grid_barrier(bar_cnt, bar_gen);
    hprev = hnext;
  }
}

// ------------------------------------------------------ decoder GEMM (WMMA) ---
// per block: one timestep.  out[b][t][n] = h_hist[t][b][:] @ dec_W + dec_b
__global__ void __launch_bounds__(256) k_decoder(const bf16_t* __restrict__ h_hist,
                                                 const bf16_t* __restrict__ decWT,
                                                 const float* __restrict__ dec_b,
                                                 float* __restrict__ out) {
  const int t = blockIdx.x;
  const int wave = threadIdx.x >> 5, lane = threadIdx.x & 31;
  const int n = lane & 15, mb = (lane < 16) ? 0 : 8;
  const bf16_t* A0 = h_hist + (size_t)t * (cB * cH);
  for (int tid = wave; tid < 16; tid += 8) {       // 4 mtiles x 4 ntiles
    int m0 = (tid & 3) * 16, n0 = (tid >> 2) * 16;
    v8f acc = v8f_zero();
    for (int k = 0; k < cH; k += 32) {
      v16bf a = load_fragA(A0 + (size_t)m0 * cH + k, cH);
      v16bf b = load_fragB(decWT + (size_t)n0 * cH + k, cH);
      acc = wmma_bf16(a, b, acc);
    }
    float bias = dec_b[n0 + n];
#pragma unroll
    for (int r = 0; r < 8; ++r) {
      int m = m0 + mb + r;
      out[(size_t)m * (cTT * cFOUT) + (size_t)t * cFOUT + n0 + n] = acc[r] + bias;
    }
  }
}

// ------------------------------------------------------------------- launch ---

extern "C" void kernel_launch(void* const* d_in, const int* in_sizes, int n_in,
                              void* d_out, int out_size, void* d_ws, size_t ws_size,
                              hipStream_t stream) {
  (void)in_sizes; (void)n_in; (void)out_size; (void)ws_size;
  const float* input_seq = (const float*)d_in[0];
  // d_in[1] = future_n (256, compile-time constant here)
  const float* enc_W = (const float*)d_in[2];
  const float* enc_b = (const float*)d_in[3];
  const float* W_ih  = (const float*)d_in[4];
  const float* W_hh  = (const float*)d_in[5];
  const float* b_ih  = (const float*)d_in[6];
  const float* b_hh  = (const float*)d_in[7];
  const float* fut_W = (const float*)d_in[8];
  const float* fut_b = (const float*)d_in[9];
  const float* dec_W = (const float*)d_in[10];
  const float* dec_b = (const float*)d_in[11];
  float* out = (float*)d_out;

  char* base = (char*)d_ws;
  size_t off = 0;
  auto take = [&](size_t bytes) -> char* {
    char* p = base + off;
    off = (off + bytes + 255) & ~(size_t)255;
    return p;
  };
  bf16_t* Wih_bf  = (bf16_t*)take((size_t)cG * cRI * 2);      // 1 MB
  bf16_t* Whh_bf  = (bf16_t*)take((size_t)cG * cH * 2);       // 2 MB
  bf16_t* Wcmb_bf = (bf16_t*)take((size_t)cG * cH * 2);       // 2 MB
  bf16_t* encWT   = (bf16_t*)take((size_t)cRI * cFIN * 2);
  bf16_t* decWT   = (bf16_t*)take((size_t)cFOUT * cH * 2);
  float*  bias_p  = (float*)take((size_t)cG * 4);
  float*  bias_f  = (float*)take((size_t)cG * 4);
  bf16_t* in_bf   = (bf16_t*)take((size_t)cT * cB * cFIN * 2);   // 8 MB
  bf16_t* enc_bf  = (bf16_t*)take((size_t)cT * cB * cRI * 2);    // 32 MB
  bf16_t* h0_bf   = (bf16_t*)take((size_t)cB * cH * 2);
  bf16_t* h_hist  = (bf16_t*)take((size_t)cTT * cB * cH * 2);    // 80 MB
  int*    bar     = (int*)take(256);

  const int TPB = 256;
  auto blocks = [](int n) { return (n + 255) / 256; };

  // allow 256 KB dynamic LDS for the persistent recurrent kernel
  (void)hipFuncSetAttribute((const void*)k_recurrent,
                            hipFuncAttributeMaxDynamicSharedMemorySize, LDS_TOTAL);

  // --- prep (runs every call: deterministic) ---
  k_zero_i32 <<<1, 32, 0, stream>>>(bar, 8);
  k_cast_bf16<<<blocks(cG * cRI), TPB, 0, stream>>>(W_ih, Wih_bf, cG * cRI);
  k_cast_bf16<<<blocks(cG * cH),  TPB, 0, stream>>>(W_hh, Whh_bf, cG * cH);
  k_tr_cast  <<<blocks(cFIN * cRI), TPB, 0, stream>>>(enc_W, encWT, cFIN, cRI);
  k_tr_cast  <<<blocks(cH * cFOUT), TPB, 0, stream>>>(dec_W, decWT, cH, cFOUT);
  k_in_tr    <<<blocks(cB * cT * cFIN), TPB, 0, stream>>>(input_seq, in_bf);
  k_fill_bf16<<<blocks(cB * cH), TPB, 0, stream>>>(h0_bf, cB * cH, 0.01f);
  k_bias     <<<blocks(cG), TPB, 0, stream>>>(b_ih, b_hh, W_ih, fut_b, bias_p, bias_f);
  k_comb     <<<blocks(cG * cH), TPB, 0, stream>>>(W_hh, W_ih, fut_W, Wcmb_bf);

  // --- parallel encoder GEMM ---
  k_encoder<<<cT, TPB, 0, stream>>>(in_bf, encWT, enc_b, enc_bf);

  // --- sequential LSTM recurrence (persistent grid, per-step global barrier) ---
  k_recurrent<<<NWG, TPB, LDS_TOTAL, stream>>>(enc_bf, h0_bf, Wih_bf, Whh_bf,
                                               Wcmb_bf, bias_p, bias_f, h_hist,
                                               bar, bar + 1);

  // --- parallel decoder GEMM ---
  k_decoder<<<cTT, TPB, 0, stream>>>(h_hist, decWT, dec_b, out);
}
